// LevyTokenFoam_83794811945128
// MI455X (gfx1250) — matrix operand order repaired and checked by reading, not compiled, not run on hardware
//
#include <hip/hip_runtime.h>

// ---------------------------------------------------------------------------
// LevyTokenFoam for MI455X (gfx1250, wave32).
// Rank-12 factorization: rho_mix = (1/12) M^T M, M = 12 normalized bubble
// states. logits[r] = (1/12) sum_j (E_r . v_j)^2  -> tall-skinny bf16 WMMA
// GEMM (32000x128 @ 128x16). Eigenvalues of rho via 12x12 Gram + Jacobi.
// ---------------------------------------------------------------------------

typedef __attribute__((ext_vector_type(16))) __bf16 v16bf;
typedef __attribute__((ext_vector_type(8)))  float  v8f;

#define T_STEPS 48
#define VOCAB   32000
#define D       128
#define NB      4
#define KF      3
#define NSTEPS  5
#define RANK    12          // KF * NB

// ------------------------- block reductions --------------------------------
__device__ __forceinline__ float bsum(float v, float* red) {
#pragma unroll
  for (int o = 16; o > 0; o >>= 1) v += __shfl_xor(v, o, 32);
  const int w = threadIdx.x >> 5;
  if ((threadIdx.x & 31) == 0) red[w] = v;
  __syncthreads();
  if (threadIdx.x == 0) {
    float s = 0.f;
    const int nw = blockDim.x >> 5;
    for (int i = 0; i < nw; ++i) s += red[i];
    red[0] = s;
  }
  __syncthreads();
  float s = red[0];
  __syncthreads();
  return s;
}

__device__ __forceinline__ float bmax(float v, float* red) {
#pragma unroll
  for (int o = 16; o > 0; o >>= 1) v = fmaxf(v, __shfl_xor(v, o, 32));
  const int w = threadIdx.x >> 5;
  if ((threadIdx.x & 31) == 0) red[w] = v;
  __syncthreads();
  if (threadIdx.x == 0) {
    float s = -3.4e38f;
    const int nw = blockDim.x >> 5;
    for (int i = 0; i < nw; ++i) s = fmaxf(s, red[i]);
    red[0] = s;
  }
  __syncthreads();
  float s = red[0];
  __syncthreads();
  return s;
}

// ------------------------- E -> bf16 ---------------------------------------
__global__ void __launch_bounds__(256) cvt_kernel(const float* __restrict__ E,
                                                  __bf16* __restrict__ Ebf, int n) {
  int i = (blockIdx.x * blockDim.x + threadIdx.x) * 8;
  if (i >= n) return;
#pragma unroll
  for (int j = 0; j < 8; ++j) Ebf[i + j] = (__bf16)E[i + j];
}

// ------------------------- foam step (1 block, 128 thr) --------------------
__global__ void __launch_bounds__(128) foam_kernel(
    const int* __restrict__ tokens, const float* __restrict__ E,
    const float* __restrict__ bases, const float* __restrict__ noise,
    const float* __restrict__ temp, const float* __restrict__ target,
    const float* __restrict__ stepsz, const float* __restrict__ modul,
    const float* __restrict__ gatel, const float* __restrict__ mdbp,
    const float* __restrict__ sensp, float* __restrict__ memw,
    float* __restrict__ interw, __bf16* __restrict__ Vbf,
    float* __restrict__ rho_out, float* __restrict__ S_out, int t) {
  __shared__ float sRed[32];
  __shared__ float sV[RANK][D];
  __shared__ float sSt[NB][D];
  __shared__ float sEx[NB][D];
  __shared__ float sFs[NB][D];
  __shared__ float sX[D];
  __shared__ float sInter[16];
  __shared__ float sG[RANK * RANK];
  const int d = threadIdx.x;

  if (t == 0) {
    // re-initialize per-launch persistent state (deterministic across replays)
    for (int i = d; i < KF * NB * D; i += D) memw[i] = 0.f;
    // interaction = softmax(-tension/temp) depends only on bases: compute once
    for (int k = 0; k < KF; ++k) {
      float tn[16];
      for (int i = 0; i < NB; ++i) tn[i * 4 + i] = sqrtf(1e-12f);
      for (int i = 0; i < NB; ++i)
        for (int j = i + 1; j < NB; ++j) {
          const float* Bi = bases + (size_t)(k * NB + i) * D * D;
          const float* Bj = bases + (size_t)(k * NB + j) * D * D;
          float p = 0.f;
          for (int e = d; e < D * D; e += D) { float df = Bi[e] - Bj[e]; p += df * df; }
          float tv = sqrtf(bsum(p, sRed) + 1e-12f);
          tn[i * 4 + j] = tv; tn[j * 4 + i] = tv;
        }
      const float tc = fmaxf(fabsf(temp[k]), 0.01f);
      if (d < 16) {
        const int i = d >> 2;
        float num = expf(-tn[d] / tc);
        float den = 0.f;
        for (int j = 0; j < NB; ++j) den += expf(-tn[i * 4 + j] / tc);
        interw[k * 16 + d] = num / den;
      }
      __syncthreads();
    }
  }

  const int tok = tokens[t];
  const float gate = 1.f / (1.f + expf(-gatel[0]));
  const float mdb = mdbp[0];
  const float sens = fabsf(sensp[0]);
  const float x = E[(size_t)tok * D + d];

  for (int k = 0; k < KF; ++k) {
    if (d < 16) sInter[d] = interw[k * 16 + d];
    float mold[NB];
    float mm = 0.f;
    for (int n = 0; n < NB; ++n) { mold[n] = memw[(k * NB + n) * D + d]; mm += mold[n]; }
    mm *= 0.25f;
    const float mem_norm = sqrtf(bsum(mm * mm, sRed)) + 1e-10f;
    const float x_norm   = sqrtf(bsum(x * x, sRed)) + 1e-10f;
    const float dxm      = bsum(x * mm, sRed);
    const float novelty  = (mem_norm > 1e-8f) ? (1.f - dxm / (x_norm * mem_norm)) : 1.f;
    const float decay    = 1.f / (1.f + expf(-(mdb - sens * novelty)));
    const float xwm = x + decay * mm;
    sX[d] = xwm;
    const float state_scale = sqrtf(bsum(xwm * xwm, sRed)) + 1e-10f;  // also syncs sX

    // measurements[n][d] = sum_c xwm[c] * bases[k][n][c][d]
    const float* Bk = bases + (size_t)k * NB * D * D;
    for (int n = 0; n < NB; ++n) {
      const float* Bn = Bk + (size_t)n * D * D;
      float acc = 0.f;
      for (int c = 0; c < D; ++c) acc += sX[c] * Bn[c * D + d];
      sSt[n][d] = acc;
    }
    const float bstep = fminf(fmaxf(fabsf(stepsz[k]), 0.001f), 0.5f);
    const float modc  = fminf(fmaxf(fabsf(modul[k]), 0.01f), 5.f);
    const float tgt   = target[k];
    __syncthreads();

    for (int it = 0; it < NSTEPS; ++it) {
      // expr[n][kk] = sum_c state[n][c] * bases[n][kk][c]   (thread = kk)
      for (int n = 0; n < NB; ++n) {
        const float* row = Bk + (size_t)n * D * D + (size_t)d * D;
        float acc = 0.f;
        for (int c = 0; c < D; ++c) acc += sSt[n][c] * row[c];
        sEx[n][d] = acc;
      }
      __syncthreads();
      float en[NB];
      for (int n = 0; n < NB; ++n)
        en[n] = sqrtf(bsum(sEx[n][d] * sEx[n][d], sRed)) + 1e-10f;
      float cosv[16];
      for (int i = 0; i < NB; ++i)
        for (int j = i + 1; j < NB; ++j) {
          float cv = bsum(sEx[i][d] * sEx[j][d], sRed) / (en[i] * en[j]);
          cosv[i * 4 + j] = cv; cosv[j * 4 + i] = cv;
        }
      float agree = 0.f;
      for (int i = 0; i < NB; ++i)
        for (int j = 0; j < NB; ++j) if (i != j) agree += cosv[i * 4 + j];
      agree *= (1.f / 12.f);
      const float unc = fmaxf(1.f - agree, 0.1f);
      const float stepv = bstep * powf(unc, modc);
      float dnv[16];
      for (int i = 0; i < NB; ++i)
        for (int j = i + 1; j < NB; ++j) {
          float df = sEx[i][d] - sEx[j][d];
          float s = sqrtf(bsum(df * df, sRed));
          dnv[i * 4 + j] = s; dnv[j * 4 + i] = s;
        }
      for (int i = 0; i < NB; ++i) {
        float acc = 0.f;
        for (int j = 0; j < NB; ++j) {
          if (j == i) continue;  // mask
          float fm = (cosv[i * 4 + j] - tgt) * sInter[i * 4 + j];
          acc += fm * (sEx[i][d] - sEx[j][d]) / (dnv[i * 4 + j] + 1e-10f);
        }
        sFs[i][d] = acc;
      }
      __syncthreads();
      // forces_local[n][d] = sum_kk fs[n][kk] * bases[n][kk][d]
      for (int n = 0; n < NB; ++n) {
        const float* Bn = Bk + (size_t)n * D * D;
        float acc = 0.f;
        for (int kk = 0; kk < D; ++kk) acc += sFs[n][kk] * Bn[kk * D + d];
        sSt[n][d] += stepv * acc;
      }
      __syncthreads();
    }

    // noise, memory update, normalized bubble vectors
    for (int n = 0; n < NB; ++n) {
      const float nz = noise[(((size_t)t * KF + k) * NB + n) * D + d];
      const float eq = sSt[n][d] + gate * nz * state_scale * 0.01f;
      memw[(k * NB + n) * D + d] = decay * mold[n] + (1.f - decay) * eq;
      const float nn = sqrtf(bsum(eq * eq, sRed)) + 1e-10f;
      const float vn = eq / nn;
      sV[k * NB + n][d] = vn;
      Vbf[(k * NB + n) * D + d] = (__bf16)vn;
    }
    __syncthreads();
  }

  // pad WMMA B-matrix rows 12..15 with zeros
  for (int j = RANK; j < 16; ++j) Vbf[j * D + d] = (__bf16)0.f;

  // rho_mix = (1/12) sum_j v_j v_j^T  (write to output)
  for (int r = 0; r < D; ++r) {
    float acc = 0.f;
#pragma unroll
    for (int j = 0; j < RANK; ++j) acc += sV[j][r] * sV[j][d];
    rho_out[r * D + d] = acc * (1.f / 12.f);
  }

  // 12x12 Gram: nonzero eigenvalues of rho_mix
  for (int idx = d; idx < RANK * RANK; idx += D) {
    const int j = idx / RANK, l = idx % RANK;
    float acc = 0.f;
    for (int c = 0; c < D; ++c) acc += sV[j][c] * sV[l][c];
    sG[idx] = acc * (1.f / 12.f);
  }
  __syncthreads();

  if (d == 0) {
    float A[RANK][RANK];
    for (int i = 0; i < RANK; ++i)
      for (int j = 0; j < RANK; ++j) A[i][j] = sG[i * RANK + j];
    for (int sw = 0; sw < 12; ++sw)
      for (int p = 0; p < RANK - 1; ++p)
        for (int q = p + 1; q < RANK; ++q) {
          const float apq = A[p][q];
          if (fabsf(apq) < 1e-13f) continue;
          const float theta = 0.5f * (A[q][q] - A[p][p]) / apq;
          const float tt = (theta >= 0.f ? 1.f : -1.f) /
                           (fabsf(theta) + sqrtf(theta * theta + 1.f));
          const float c = rsqrtf(tt * tt + 1.f);
          const float s = tt * c;
          for (int i = 0; i < RANK; ++i) {
            const float aip = A[i][p], aiq = A[i][q];
            A[i][p] = c * aip - s * aiq;
            A[i][q] = s * aip + c * aiq;
          }
          for (int i = 0; i < RANK; ++i) {
            const float api = A[p][i], aqi = A[q][i];
            A[p][i] = c * api - s * aqi;
            A[q][i] = s * api + c * aqi;
          }
        }
    float tot = 116.f * 1e-12f;  // 116 zero-eigs clamped to 1e-12
    float lam[RANK];
    for (int i = 0; i < RANK; ++i) { lam[i] = fmaxf(A[i][i], 1e-12f); tot += lam[i]; }
    float S = 0.f;
    for (int i = 0; i < RANK; ++i) {
      const float p = lam[i] / tot;
      S -= p * fmaxf(logf(p), -100.f);
    }
    const float pt = 1e-12f / tot;
    S -= 116.f * pt * fmaxf(logf(pt), -100.f);
    S_out[0] = S;
  }
}

// ------------------------- WMMA logits kernel ------------------------------
// A fragment (16x32 bf16): lane<16 -> row=lane, K in {kb..kb+7} u {kb+16..kb+23}
//                          lane>=16 -> row=lane-16, K offset +8.
__device__ __forceinline__ v16bf loadA(const __bf16* __restrict__ M, int row,
                                       int kbase, int lane) {
  const __bf16* p = M + (size_t)row * D + kbase + ((lane & 16) ? 8 : 0);
  union { uint4 u; __bf16 h[8]; } a0, a1;
  a0.u = *(const uint4*)p;
  a1.u = *(const uint4*)(p + 16);
  v16bf f;
#pragma unroll
  for (int e = 0; e < 8; ++e) { f[e] = a0.h[e]; f[8 + e] = a1.h[e]; }
  return f;
}

// B fragment (32x16 bf16): lane holds column N=lane&15; K in {kb..kb+7} u {kb+16..kb+23}, +8 for lane>=16
__device__ __forceinline__ v16bf loadB(const __bf16* __restrict__ Vt, int kbase,
                                       int lane) {
  const __bf16* p = Vt + (size_t)(lane & 15) * D + kbase + ((lane & 16) ? 8 : 0);
  union { uint4 u; __bf16 h[8]; } a0, a1;
  a0.u = *(const uint4*)p;
  a1.u = *(const uint4*)(p + 16);
  v16bf f;
#pragma unroll
  for (int e = 0; e < 8; ++e) { f[e] = a0.h[e]; f[8 + e] = a1.h[e]; }
  return f;
}

__global__ void __launch_bounds__(256) logits_kernel(
    const __bf16* __restrict__ Ebf, const __bf16* __restrict__ Vbf,
    float* __restrict__ logits) {
  const int lane = threadIdx.x & 31;
  const int wave = threadIdx.x >> 5;
  const int tile = blockIdx.x * 8 + wave;     // 250 blocks * 8 waves = 2000 tiles
  const int rowbase = tile * 16;
  const int arow = rowbase + (lane & 15);

  const v16bf b0 = loadB(Vbf, 0, lane);
  const v16bf b1 = loadB(Vbf, 32, lane);
  const v16bf b2 = loadB(Vbf, 64, lane);
  const v16bf b3 = loadB(Vbf, 96, lane);

  v8f c = {};
  v16bf a;
  a = loadA(Ebf, arow, 0, lane);
  c = __builtin_amdgcn_wmma_f32_16x16x32_bf16(false, a, false, b0, (short)0, c, false, false);
  a = loadA(Ebf, arow, 32, lane);
  c = __builtin_amdgcn_wmma_f32_16x16x32_bf16(false, a, false, b1, (short)0, c, false, false);
  a = loadA(Ebf, arow, 64, lane);
  c = __builtin_amdgcn_wmma_f32_16x16x32_bf16(false, a, false, b2, (short)0, c, false, false);
  a = loadA(Ebf, arow, 96, lane);
  c = __builtin_amdgcn_wmma_f32_16x16x32_bf16(false, a, false, b3, (short)0, c, false, false);

  // logits[row] = (1/12) * sum_{j<12} C[row][j]^2 ; reduce across 16-lane half
  float out[8];
#pragma unroll
  for (int p = 0; p < 8; ++p) {
    float v = c[p];
    float sq = ((lane & 15) < RANK) ? v * v : 0.f;
#pragma unroll
    for (int m = 1; m < 16; m <<= 1) sq += __shfl_xor(sq, m, 32);
    out[p] = sq * (1.f / 12.f);
  }
  if ((lane & 15) == 0) {
    const int r0 = rowbase + ((lane & 16) ? 8 : 0);
#pragma unroll
    for (int p = 0; p < 8; ++p) logits[r0 + p] = out[p];
  }
}

// ------------------------- softmax + entropy -------------------------------
__global__ void __launch_bounds__(1024) softmax_kernel(
    const float* __restrict__ logits, float* __restrict__ probs,
    const float* __restrict__ Sin, float* __restrict__ Hout,
    float* __restrict__ HSout) {
  __shared__ float red[32];
  const int tid = threadIdx.x;
  float m = -3.4e38f;
  for (int i = tid; i < VOCAB; i += 1024) m = fmaxf(m, logits[i]);
  m = bmax(m, red);
  float z = 0.f;
  for (int i = tid; i < VOCAB; i += 1024) z += expf(logits[i] - m);
  z = bsum(z, red);
  const float lz = logf(z);
  float h = 0.f;
  for (int i = tid; i < VOCAB; i += 1024) {
    const float lp = logits[i] - m - lz;
    const float p = expf(lp);
    probs[i] = p;
    h -= p * fmaxf(lp, -100.f);
  }
  h = bsum(h, red);
  if (tid == 0) { Hout[0] = h; HSout[0] = h - Sin[0]; }
}

// ------------------------- launch ------------------------------------------
extern "C" void kernel_launch(void* const* d_in, const int* in_sizes, int n_in,
                              void* d_out, int out_size, void* d_ws, size_t ws_size,
                              hipStream_t stream) {
  const int*   tokens = (const int*)d_in[0];
  const float* E      = (const float*)d_in[1];
  const float* bases  = (const float*)d_in[2];
  const float* noise  = (const float*)d_in[3];
  const float* temp   = (const float*)d_in[4];
  const float* target = (const float*)d_in[5];
  const float* stepsz = (const float*)d_in[6];
  const float* modul  = (const float*)d_in[7];
  const float* gatel  = (const float*)d_in[8];
  const float* mdb    = (const float*)d_in[9];
  const float* sens   = (const float*)d_in[10];
  float* out = (float*)d_out;

  char* w = (char*)d_ws;
  __bf16* Ebf   = (__bf16*)w; w += (size_t)VOCAB * D * sizeof(__bf16);   // 8.19 MB
  float* memw   = (float*)w;  w += (size_t)KF * NB * D * sizeof(float);  // 6 KB
  float* interw = (float*)w;  w += (size_t)KF * 16 * sizeof(float);
  __bf16* Vbf   = (__bf16*)w; w += (size_t)16 * D * sizeof(__bf16);
  float* logitsW = (float*)w; w += (size_t)VOCAB * sizeof(float);        // 128 KB

  float* probsO = out;
  float* rhoO   = out + (size_t)T_STEPS * VOCAB;
  float* SO     = rhoO + (size_t)T_STEPS * D * D;
  float* HO     = SO + T_STEPS;
  float* HSO    = HO + T_STEPS;

  cvt_kernel<<<2000, 256, 0, stream>>>(E, Ebf, VOCAB * D);
  for (int t = 0; t < T_STEPS; ++t) {
    foam_kernel<<<1, 128, 0, stream>>>(tokens, E, bases, noise, temp, target,
                                       stepsz, modul, gatel, mdb, sens, memw,
                                       interw, Vbf, rhoO + (size_t)t * D * D,
                                       SO + t, t);
    logits_kernel<<<250, 256, 0, stream>>>(Ebf, Vbf, logitsW);
    softmax_kernel<<<1, 1024, 0, stream>>>(logitsW, probsO + (size_t)t * VOCAB,
                                           SO + t, HO + t, HSO + t);
  }
}
